// SSA_rel_scl_24283745092270
// MI455X (gfx1250) — matrix operand order repaired and checked by reading, not compiled, not run on hardware
//
#include <hip/hip_runtime.h>

// ---------------------------------------------------------------------------
// SSA (spiking self-attention) for MI455X / gfx1250 — wave32, WMMA, async-LDS
// T=4 B=16 N=256 C=256 H=8 D=32 TOPK=64, tau=2, v_th=1, bn eps=1e-5
// ---------------------------------------------------------------------------

typedef __attribute__((ext_vector_type(16))) __bf16 v16bf;
typedef __attribute__((ext_vector_type(2)))  __bf16 v2bf;
typedef __attribute__((ext_vector_type(8)))  float  v8f;
typedef __attribute__((ext_vector_type(2)))  float  v2f;

union FragBF {            // 16 bf16 operand (A or B of 16x16x32 bf16 wmma)
    uint4  q[2];
    v16bf  v;
};

#if __has_builtin(__builtin_amdgcn_global_load_async_to_lds_b128)
#define HAVE_ASYNC_LDS 1
// exact param type per clang diagnostic: vector_size(16) int, AS(1)/AS(3)
typedef int vi4_t __attribute__((vector_size(16)));
typedef __attribute__((address_space(1))) vi4_t as1_vi4;
typedef __attribute__((address_space(3))) vi4_t as3_vi4;
#else
#define HAVE_ASYNC_LDS 0
#endif

// pack two f32 -> packed bf16 pair (RNE); hardware v_cvt_pk_bf16_f32 if present
__device__ inline unsigned int pack_bf2(float lo, float hi) {
#if __has_builtin(__builtin_amdgcn_cvt_pk_bf16_f32)
    v2bf r = __builtin_amdgcn_cvt_pk_bf16_f32(lo, hi);
    return __builtin_bit_cast(unsigned int, r);
#else
    unsigned int ul = __float_as_uint(lo), uh = __float_as_uint(hi);
    unsigned int rl = ul + 0x7FFFu + ((ul >> 16) & 1u);
    unsigned int rh = uh + 0x7FFFu + ((uh >> 16) & 1u);
    return (rl >> 16) | (rh & 0xFFFF0000u);
#endif
}
__device__ inline float bf2f(unsigned short s) {
    return __uint_as_float(((unsigned int)s) << 16);
}
__device__ inline unsigned int ord_u32(float f) {  // order-preserving map
    unsigned int u = __float_as_uint(f);
    return (u >> 31) ? ~u : (u | 0x80000000u);
}

// ---------------------------------------------------------------------------
// GEMM + BatchNorm(eval):  Out[m,o] = BN( sum_c A[m,c] * W[o,c] )
// M = 16384 (=T*B*N), O = C = 256, K = 256.  64x64 C-tile per block, 8 waves.
// A is f32 (x) or bf16 (spikes); staged to bf16 LDS in K=32 chunks, b128 I/O.
// ---------------------------------------------------------------------------
__global__ __launch_bounds__(256) void gemm_bn_kernel(
    const float* __restrict__ Af, const unsigned short* __restrict__ Ab,
    const float* __restrict__ W,  const float* __restrict__ bn,
    float* __restrict__ Out)
{
    __shared__ unsigned short Asm_[64 * 32];
    __shared__ unsigned short Wsm_[64 * 32];

    const int tid  = threadIdx.x;
    const int w    = tid >> 5;
    const int lane = tid & 31;
    const int half = lane >> 4;
    const int l16  = lane & 15;
    const int mBlk = blockIdx.x * 64;
    const int oBlk = blockIdx.y * 64;
    const int mt   = w >> 1;            // wave's M-tile (0..3)
    const int ot0  = (w & 1) * 2;       // wave's first O-tile (0 or 2)

    // each thread stages 8 contiguous elements of each LDS buffer
    const int sRow = tid >> 2;          // 0..63
    const int sCol = (tid & 3) * 8;     // 0,8,16,24

    v8f acc0 = {}; v8f acc1 = {};

    for (int kb = 0; kb < 256; kb += 32) {
        __syncthreads();
        {
            uint4 aPack, wPack;
            if (Af) {
                const float* ap = Af + (size_t)(mBlk + sRow) * 256 + kb + sCol;
                float4 a0 = *(const float4*)(ap);
                float4 a1 = *(const float4*)(ap + 4);
                aPack.x = pack_bf2(a0.x, a0.y); aPack.y = pack_bf2(a0.z, a0.w);
                aPack.z = pack_bf2(a1.x, a1.y); aPack.w = pack_bf2(a1.z, a1.w);
            } else {
                aPack = *(const uint4*)(Ab + (size_t)(mBlk + sRow) * 256 + kb + sCol);
            }
            const float* wp = W + (size_t)(oBlk + sRow) * 256 + kb + sCol;
            float4 w0 = *(const float4*)(wp);
            float4 w1 = *(const float4*)(wp + 4);
            wPack.x = pack_bf2(w0.x, w0.y); wPack.y = pack_bf2(w0.z, w0.w);
            wPack.z = pack_bf2(w1.x, w1.y); wPack.w = pack_bf2(w1.z, w1.w);
            ((uint4*)Asm_)[tid] = aPack;
            ((uint4*)Wsm_)[tid] = wPack;
            if (kb < 224) {   // global_prefetch_b8 of next K-chunk
                if (Af) __builtin_prefetch(Af + (size_t)(mBlk + sRow) * 256 + kb + 32 + sCol, 0, 0);
                else    __builtin_prefetch(Ab + (size_t)(mBlk + sRow) * 256 + kb + 32 + sCol, 0, 0);
                __builtin_prefetch(W + (size_t)(oBlk + sRow) * 256 + kb + 32 + sCol, 0, 0);
            }
        }
        __syncthreads();

        FragBF a;   // A 16x32 bf16 fragment: lane row = l16, K = 16*half .. +15
        {
            const unsigned short* p = &Asm_[(mt * 16 + l16) * 32];
            a.q[0] = *(const uint4*)(p + 16 * half);
            a.q[1] = *(const uint4*)(p + 16 * half + 8);
        }
        FragBF b0, b1;  // B 32x16 bf16: lane col = l16, K = 16*half .. +15
        {
            const unsigned short* p0 = &Wsm_[(ot0 * 16 + l16) * 32 + 16 * half];
            b0.q[0] = *(const uint4*)(p0);
            b0.q[1] = *(const uint4*)(p0 + 8);
            const unsigned short* p1 = &Wsm_[((ot0 + 1) * 16 + l16) * 32 + 16 * half];
            b1.q[0] = *(const uint4*)(p1);
            b1.q[1] = *(const uint4*)(p1 + 8);
        }
        acc0 = __builtin_amdgcn_wmma_f32_16x16x32_bf16(false, a.v, false, b0.v,
                                                       (short)0, acc0, false, false);
        acc1 = __builtin_amdgcn_wmma_f32_16x16x32_bf16(false, a.v, false, b1.v,
                                                       (short)0, acc1, false, false);
    }

    for (int j = 0; j < 2; ++j) {
        int oG = oBlk + (ot0 + j) * 16 + l16;
        float gam = bn[oG], bet = bn[256 + oG], mea = bn[512 + oG], var = bn[768 + oG];
        float scl = gam * rsqrtf(var + 1e-5f);
        float bia = bet - mea * scl;
        v8f acc = j ? acc1 : acc0;
        for (int r = 0; r < 8; ++r) {
            int mG = mBlk + mt * 16 + r + 8 * half;   // C-layout: M = r + 8*half
            Out[(size_t)mG * 256 + oG] = acc[r] * scl + bia;
        }
    }
}

// ---------------------------------------------------------------------------
// Multi-step LIF over T=4 (decay_input, hard reset), 4 elements per thread.
// ---------------------------------------------------------------------------
#define LIF_STEP(v, x, s)  { (v) = 0.5f * ((v) + (x));            \
                             (s) = ((v) - 1.0f) >= 0.0f;          \
                             if (s) (v) = 0.0f; }

__global__ __launch_bounds__(256) void lif_bf16_kernel(
    const float4* __restrict__ H4, unsigned short* __restrict__ S)
{
    const int i4 = blockIdx.x * 256 + threadIdx.x;   // < 262144
    float vx = 0.f, vy = 0.f, vz = 0.f, vw = 0.f;
    for (int t = 0; t < 4; ++t) {
        float4 x = H4[(size_t)t * 262144 + i4];
        bool sx, sy, sz, sw;
        LIF_STEP(vx, x.x, sx); LIF_STEP(vy, x.y, sy);
        LIF_STEP(vz, x.z, sz); LIF_STEP(vw, x.w, sw);
        uint2 o;   // 4 bf16 spikes packed as 2 dwords (0x3F80 = 1.0bf16)
        o.x = (sx ? 0x3F80u : 0u) | (sy ? 0x3F800000u : 0u);
        o.y = (sz ? 0x3F80u : 0u) | (sw ? 0x3F800000u : 0u);
        *(uint2*)(S + (size_t)t * 1048576 + (size_t)i4 * 4) = o;
    }
}

__global__ __launch_bounds__(256) void lif_f32_kernel(
    const float4* __restrict__ H4, float4* __restrict__ S4)
{
    const int i4 = blockIdx.x * 256 + threadIdx.x;
    float vx = 0.f, vy = 0.f, vz = 0.f, vw = 0.f;
    for (int t = 0; t < 4; ++t) {
        float4 x = H4[(size_t)t * 262144 + i4];
        bool sx, sy, sz, sw;
        LIF_STEP(vx, x.x, sx); LIF_STEP(vy, x.y, sy);
        LIF_STEP(vz, x.z, sz); LIF_STEP(vw, x.w, sw);
        float4 o;
        o.x = sx ? 1.f : 0.f; o.y = sy ? 1.f : 0.f;
        o.z = sz ? 1.f : 0.f; o.w = sw ? 1.f : 0.f;
        S4[(size_t)t * 262144 + i4] = o;
    }
}

// ---------------------------------------------------------------------------
// Attention: per (t,b,h) and per 32-row chunk:
//   K/V spike tiles -> LDS via GLOBAL_LOAD_ASYNC_TO_LDS_B128 (ASYNCcnt)
//   S = q k^T * 1/16 + rel_bias        (bf16 WMMA, K=D=32, one wmma per tile)
//   per-row exact top-64 threshold     (register radix select, shfl reduce)
//   out = S @ V                        (f32 WMMA 16x16x4, K=256)
// LDS: S 32x256 f32 (32KB) + K 256x32 bf16 (16KB) + V 256x32 bf16 (16KB)
// ---------------------------------------------------------------------------
__global__ __launch_bounds__(256) void attn_kernel(
    const unsigned short* __restrict__ Sq, const unsigned short* __restrict__ Sk,
    const unsigned short* __restrict__ Sv, const float* __restrict__ Tbl,
    float* __restrict__ Out)
{
    __shared__ unsigned int lds[16384];                       // 64 KB
    float*          Ssm = (float*)lds;                        // [32][256]
    unsigned short* Ksm = (unsigned short*)(lds + 8192);      // [256][32]
    unsigned short* Vsm = (unsigned short*)(lds + 12288);     // [256][32]
    unsigned int*   kthA = lds + 8192;                        // reuse K region

    const int tid  = threadIdx.x;
    const int w    = tid >> 5;
    const int lane = tid & 31;
    const int half = lane >> 4;
    const int l16  = lane & 15;
    const int rc   = blockIdx.x;        // row chunk 0..7 (32 rows each)
    const int tbh  = blockIdx.y;        // 0..511
    const int hh   = tbh & 7;
    const int tb   = tbh >> 3;          // t*16 + b
    const size_t baseTB = (size_t)tb * 256 * 256;

    // --- stage K, V head tiles (row n contiguous over d, 64B per row) ---
    for (int i = tid; i < 1024; i += 256) {
        int row = i >> 2, q4 = i & 3;
        const unsigned short* gk = Sk + baseTB + (size_t)row * 256 + hh * 32 + q4 * 8;
        const unsigned short* gv = Sv + baseTB + (size_t)row * 256 + hh * 32 + q4 * 8;
#if HAVE_ASYNC_LDS
        __builtin_amdgcn_global_load_async_to_lds_b128(
            (as1_vi4*)gk, (as3_vi4*)(Ksm + (size_t)i * 8), 0, 0);
        __builtin_amdgcn_global_load_async_to_lds_b128(
            (as1_vi4*)gv, (as3_vi4*)(Vsm + (size_t)i * 8), 0, 0);
#else
        ((uint4*)Ksm)[i] = *(const uint4*)gk;
        ((uint4*)Vsm)[i] = *(const uint4*)gv;
#endif
    }
#if HAVE_ASYNC_LDS
#if __has_builtin(__builtin_amdgcn_s_wait_asynccnt)
    __builtin_amdgcn_s_wait_asynccnt(0);
#else
    asm volatile("s_wait_asynccnt 0x0" ::: "memory");
#endif
#endif
    __syncthreads();

    // --- S = q k^T * scale + rel_bias ---
    const int mt = w >> 2;                       // 0..1
    FragBF a;
    {
        int nq = rc * 32 + mt * 16 + l16;
        const unsigned short* p = Sq + baseTB + (size_t)nq * 256 + hh * 32;
        a.q[0] = *(const uint4*)(p + 16 * half);
        a.q[1] = *(const uint4*)(p + 16 * half + 8);
    }
    for (int i = 0; i < 4; ++i) {
        int nt = (w & 3) * 4 + i;                // 0..15
        FragBF bfr;
        const unsigned short* pb = &Ksm[(nt * 16 + l16) * 32 + 16 * half];
        bfr.q[0] = *(const uint4*)(pb);
        bfr.q[1] = *(const uint4*)(pb + 8);
        v8f acc = {};
        acc = __builtin_amdgcn_wmma_f32_16x16x32_bf16(false, a.v, false, bfr.v,
                                                      (short)0, acc, false, false);
        for (int r = 0; r < 8; ++r) {
            int mloc = mt * 16 + r + 8 * half;
            int iAbs = rc * 32 + mloc;
            int jAbs = nt * 16 + l16;
            float bias = Tbl[(size_t)(jAbs - iAbs + 255) * 8 + hh];
            Ssm[mloc * 256 + jAbs] = acc[r] * 0.0625f + bias;
        }
    }
    __syncthreads();

    // --- exact top-64 threshold per row: register radix select, 8 lanes/row ---
    {
        const int row = tid >> 3, sub = tid & 7;
        unsigned int rv[32];
        const float4* rp = (const float4*)&Ssm[row * 256 + sub * 32];
#pragma unroll
        for (int j = 0; j < 8; ++j) {
            float4 f = rp[j];
            rv[4 * j + 0] = ord_u32(f.x); rv[4 * j + 1] = ord_u32(f.y);
            rv[4 * j + 2] = ord_u32(f.z); rv[4 * j + 3] = ord_u32(f.w);
        }
        unsigned int thr = 0;
        for (int bit = 31; bit >= 0; --bit) {
            unsigned int cand = thr | (1u << bit);
            int cnt = 0;
#pragma unroll
            for (int j = 0; j < 32; ++j) cnt += (rv[j] >= cand) ? 1 : 0;
            cnt += __shfl_xor(cnt, 1, 8);
            cnt += __shfl_xor(cnt, 2, 8);
            cnt += __shfl_xor(cnt, 4, 8);
            if (cnt >= 64) thr = cand;           // TOPK = 64
        }
        if (sub == 0) kthA[row] = thr;           // == ord(64th largest), exact
    }
    __syncthreads();
    for (int e = tid; e < 8192; e += 256) {
        int row = e >> 8;
        if (ord_u32(Ssm[e]) < kthA[row]) Ssm[e] = 0.0f;
    }
    __syncthreads();

    // --- out = S @ V  (f32 wmma 16x16x4, K = 256) ---
    if (w < 4) {
        const int mt2 = w >> 1, dt = w & 1;
        v8f acc = {};
        for (int kb = 0; kb < 256; kb += 4) {
            int krow = kb + 2 * half;
            v2f aa = *(const v2f*)&Ssm[(mt2 * 16 + l16) * 256 + krow];
            v2f bb;
            bb.x = bf2f(Vsm[(krow    ) * 32 + dt * 16 + l16]);
            bb.y = bf2f(Vsm[(krow + 1) * 32 + dt * 16 + l16]);
            acc = __builtin_amdgcn_wmma_f32_16x16x4_f32(false, aa, false, bb,
                                                        (short)0, acc, false, false);
        }
        for (int r = 0; r < 8; ++r) {
            int nAbs = rc * 32 + mt2 * 16 + r + 8 * half;
            Out[baseTB + (size_t)nAbs * 256 + hh * 32 + dt * 16 + l16] = acc[r];
        }
    }
}

// ---------------------------------------------------------------------------
// Launch: gemm+bn -> lif (x3 for q,k,v) -> attn -> lif -> gemm+bn -> lif
// Workspace: h f32 (16.78MB) + 4 spike buffers bf16 (8.39MB each) = ~50.3MB
// ---------------------------------------------------------------------------
extern "C" void kernel_launch(void* const* d_in, const int* in_sizes, int n_in,
                              void* d_out, int out_size, void* d_ws, size_t ws_size,
                              hipStream_t stream) {
    (void)in_sizes; (void)n_in; (void)out_size; (void)ws_size;
    const float* x   = (const float*)d_in[0];
    const float* wq  = (const float*)d_in[1];
    const float* wk  = (const float*)d_in[2];
    const float* wv  = (const float*)d_in[3];
    const float* bnq = (const float*)d_in[4];
    const float* bnk = (const float*)d_in[5];
    const float* bnv = (const float*)d_in[6];
    const float* pw  = (const float*)d_in[7];
    const float* pbn = (const float*)d_in[8];
    const float* tbl = (const float*)d_in[9];

    char* ws = (char*)d_ws;
    float* h = (float*)ws;                                     // [T,B,N,C] f32
    unsigned short* sq = (unsigned short*)(ws + (size_t)16777216);
    unsigned short* sk = sq + 4194304;
    unsigned short* sv = sk + 4194304;
    unsigned short* s2 = sv + 4194304;

    dim3 gG(256, 4), blk(256);
    dim3 gA(8, 512);

    gemm_bn_kernel<<<gG, blk, 0, stream>>>(x, nullptr, wq, bnq, h);
    lif_bf16_kernel<<<1024, blk, 0, stream>>>((const float4*)h, sq);
    gemm_bn_kernel<<<gG, blk, 0, stream>>>(x, nullptr, wk, bnk, h);
    lif_bf16_kernel<<<1024, blk, 0, stream>>>((const float4*)h, sk);
    gemm_bn_kernel<<<gG, blk, 0, stream>>>(x, nullptr, wv, bnv, h);
    lif_bf16_kernel<<<1024, blk, 0, stream>>>((const float4*)h, sv);

    attn_kernel<<<gA, blk, 0, stream>>>(sq, sk, sv, tbl, h);
    lif_bf16_kernel<<<1024, blk, 0, stream>>>((const float4*)h, s2);

    gemm_bn_kernel<<<gG, blk, 0, stream>>>(nullptr, s2, pw, pbn, h);
    lif_f32_kernel<<<1024, blk, 0, stream>>>((const float4*)h, (float4*)d_out);
}
